// MixtureAttention_8890582302736
// MI455X (gfx1250) — matrix operand order repaired and checked by reading, not compiled, not run on hardware
//
#include <hip/hip_runtime.h>
#include <math.h>

typedef __bf16 bf16_t;
typedef __attribute__((ext_vector_type(16))) __bf16 v16bf;
typedef __attribute__((ext_vector_type(8)))  __bf16 v8bf;
typedef __attribute__((ext_vector_type(8)))  float  v8f;

namespace {
constexpr int H  = 512, B = 8, L = 128, A = 50;
constexpr int VN = 1000;
constexpr int K3H = 1536;            // 3*H
constexpr int NG  = 10003;           // VT + 3
constexpr int NPAD = 10112;          // 79 * 128
constexpr int GEMM_BLOCKS = NPAD / 128;

constexpr size_t alignup(size_t x) { return (x + 255) & ~(size_t)255; }
constexpr size_t OFF_WGT    = 0;                                                 // bf16 [NPAD][K3H] (W_g transposed)
constexpr size_t OFF_WIH    = alignup(OFF_WGT  + (size_t)NPAD * K3H * 2);        // bf16 [2048][512]
constexpr size_t OFF_WHH    = alignup(OFF_WIH  + (size_t)2048 * 512 * 2);        // bf16 [2048][512]
constexpr size_t OFF_WHID   = alignup(OFF_WHH  + (size_t)2048 * 512 * 2);        // bf16 [512][512]
constexpr size_t OFF_WMEM   = alignup(OFF_WHID + (size_t)512 * 512 * 2);         // bf16 [512][512]
// ---- zero-initialized span starts here ----
constexpr size_t OFF_HS     = alignup(OFF_WMEM + (size_t)512 * 512 * 2);         // f32 [L][B][H]
constexpr size_t OFF_HSW    = alignup(OFF_HS   + (size_t)L * B * H * 4);         // f32 [L][B][H]  (W_mem@h + b_mem)
constexpr size_t OFF_H      = alignup(OFF_HSW  + (size_t)L * B * H * 4);         // f32 [B][H]
constexpr size_t OFF_C      = alignup(OFF_H    + (size_t)B * H * 4);             // f32 [B][H]
constexpr size_t OFF_ZBUF   = alignup(OFF_C    + (size_t)B * H * 4);             // bf16 [16][K3H] (rows 8..15 stay 0)
// ---- zero span ends here ----
constexpr size_t OFF_GATES  = alignup(OFF_ZBUF + (size_t)16 * K3H * 2);          // f32 [B][2048]
constexpr size_t OFF_WHB    = alignup(OFF_GATES+ (size_t)B * 2048 * 4);          // f32 [B][H]   (W_hid@h + b_hid)
constexpr size_t OFF_LOGITS = alignup(OFF_WHB  + (size_t)B * H * 4);             // f32 [B][NPAD]
constexpr size_t OFF_ST     = alignup(OFF_LOGITS + (size_t)B * NPAD * 4);        // f32 [B]
constexpr size_t OFF_LOGATT = alignup(OFF_ST   + 256);                           // f32 [B][64]
constexpr size_t OFF_LOSS   = alignup(OFF_LOGATT + (size_t)B * 64 * 4);          // f32 [L][B]
} // namespace

__device__ __forceinline__ float sigf(float x) { return 1.f / (1.f + expf(-x)); }
__device__ __forceinline__ float wred_sum(float v) {
  for (int o = 16; o; o >>= 1) v += __shfl_xor(v, o, 32);
  return v;
}
__device__ __forceinline__ float wred_max(float v) {
  for (int o = 16; o; o >>= 1) v = fmaxf(v, __shfl_xor(v, o, 32));
  return v;
}
__device__ __forceinline__ v8f wmma_bf16(v16bf a, v16bf b, v8f c) {
  return __builtin_amdgcn_wmma_f32_16x16x32_bf16(false, a, false, b, (short)0, c, false, false);
}
__device__ __forceinline__ v16bf mk16(const bf16_t* p) {
  // A-fragment halves: elements 0..7 at p, elements 8..15 at p+8 (16B-aligned each)
  v8bf lo = *(const v8bf*)p;
  v8bf hi = *(const v8bf*)(p + 8);
  return __builtin_shufflevector(lo, hi, 0,1,2,3,4,5,6,7,8,9,10,11,12,13,14,15);
}

// ------------------------- setup kernels -------------------------

__global__ void k_zero(char* __restrict__ ws) {
  float* p = (float*)(ws + OFF_HS);
  size_t n = (OFF_GATES - OFF_HS) / 4;
  for (size_t i = (size_t)blockIdx.x * blockDim.x + threadIdx.x; i < n;
       i += (size_t)gridDim.x * blockDim.x)
    p[i] = 0.f;
}

__global__ void k_cvt_small(char* __restrict__ ws,
                            const float* __restrict__ wih, const float* __restrict__ whh,
                            const float* __restrict__ whid, const float* __restrict__ wmem) {
  bf16_t* dih  = (bf16_t*)(ws + OFF_WIH);
  bf16_t* dhh  = (bf16_t*)(ws + OFF_WHH);
  bf16_t* dhid = (bf16_t*)(ws + OFF_WHID);
  bf16_t* dmem = (bf16_t*)(ws + OFF_WMEM);
  const size_t N1 = (size_t)2048 * 512, N2 = (size_t)512 * 512;
  const size_t total = 2 * N1 + 2 * N2;
  for (size_t i = (size_t)blockIdx.x * blockDim.x + threadIdx.x; i < total;
       i += (size_t)gridDim.x * blockDim.x) {
    if (i < N1)                dih[i]            = (bf16_t)wih[i];
    else if (i < 2 * N1)       dhh[i - N1]       = (bf16_t)whh[i - N1];
    else if (i < 2 * N1 + N2)  dhid[i - 2 * N1]  = (bf16_t)whid[i - 2 * N1];
    else                       dmem[i - 2*N1-N2] = (bf16_t)wmem[i - 2 * N1 - N2];
  }
}

__global__ void k_cvt_wg(char* __restrict__ ws, const float* __restrict__ wg) {
  const int n = blockIdx.x; // 0..NPAD-1
  bf16_t* dst = (bf16_t*)(ws + OFF_WGT) + (size_t)n * K3H;
  for (int k = threadIdx.x; k < K3H; k += blockDim.x)
    dst[k] = (n < NG) ? (bf16_t)wg[(size_t)k * NG + n] : (bf16_t)0.f;
}

// ------------------------- per-step kernels -------------------------

__global__ void __launch_bounds__(256)
k_step(char* __restrict__ ws, int t,
       const float* __restrict__ embN, const float* __restrict__ embT,
       const int* __restrict__ n_tensor, const int* __restrict__ t_tensor,
       const int* __restrict__ p_tensor,
       const float* __restrict__ b_ih, const float* __restrict__ b_hh,
       const float* __restrict__ b_hid, const float* __restrict__ b_mem,
       const float* __restrict__ v_w, const float* __restrict__ v_b,
       const float* __restrict__ W_s, const float* __restrict__ b_s) {
  __shared__ bf16_t ldsX[16 * H];
  __shared__ bf16_t ldsH[16 * H];
  __shared__ float  ldsCtx[B * H];
  __shared__ float  ldsSc[B * 64];
  __shared__ float  ldsAttn[B * 64];
  __shared__ int    ldsIdx[24];

  const int tid = threadIdx.x;
  const int wv = tid >> 5, lane = tid & 31;

  const bf16_t* Wih  = (const bf16_t*)(ws + OFF_WIH);
  const bf16_t* Whh  = (const bf16_t*)(ws + OFF_WHH);
  const bf16_t* Whid = (const bf16_t*)(ws + OFF_WHID);
  const bf16_t* Wmem = (const bf16_t*)(ws + OFF_WMEM);
  float*  hs    = (float*)(ws + OFF_HS);
  float*  hsW   = (float*)(ws + OFF_HSW);
  float*  hws   = (float*)(ws + OFF_H);
  float*  cws   = (float*)(ws + OFF_C);
  bf16_t* zbuf  = (bf16_t*)(ws + OFF_ZBUF);
  float*  gates = (float*)(ws + OFF_GATES);
  float*  Whb   = (float*)(ws + OFF_WHB);
  float*  stws  = (float*)(ws + OFF_ST);
  float*  lgat  = (float*)(ws + OFF_LOGATT);

  // P0: teacher-forced indices + stage x, h_prev in LDS as bf16 (rows 8..15 zero)
  if (tid < 8) {
    ldsIdx[tid]      = (t == 0) ? 0 : n_tensor[tid * L + t - 1];
    ldsIdx[8 + tid]  = (t == 0) ? 0 : t_tensor[tid * L + t - 1];
    ldsIdx[16 + tid] = (t == 0) ? 0 : p_tensor[tid * L + t - 1];
  }
  for (int i = tid; i < 8 * H; i += 256) {
    ldsX[8 * H + i] = (bf16_t)0.f;
    ldsH[8 * H + i] = (bf16_t)0.f;
  }
  __syncthreads();
  for (int i = tid; i < 8 * H; i += 256) {
    int b = i >> 9, k = i & (H - 1);
    float xv = (k < 256) ? embN[(size_t)ldsIdx[b] * 256 + k]
                         : embT[(size_t)ldsIdx[8 + b] * 256 + (k - 256)];
    ldsX[b * H + k] = (bf16_t)xv;
    ldsH[b * H + k] = (bf16_t)hws[b * H + k];
  }
  __syncthreads();

  const int koffA = (lane < 16) ? 0 : 16;  // A-fragment K offset (elements)
  const int koffB = (lane < 16) ? 0 : 16;  // B-fragment K offset (elements)
  const int rowA  = lane & 15;

  // P1: gates = x@W_ih^T + h@W_hh^T + (b_ih+b_hh)   [16x512]x[512x2048]
  for (int strip = wv; strip < 128; strip += 8) {
    int n = strip * 16 + (lane & 15);
    v8f acc; for (int r = 0; r < 8; r++) acc[r] = 0.f;
    {
      const bf16_t* pb = Wih + (size_t)n * H + koffB;
      v16bf bcur = *(const v16bf*)pb;
      #pragma unroll
      for (int kb = 0; kb < H; kb += 32) {
        v16bf bnx = bcur;
        if (kb + 32 < H) bnx = *(const v16bf*)(pb + kb + 32);  // prefetch next B
        v16bf a = mk16(&ldsX[rowA * H + kb + koffA]);
        acc = wmma_bf16(a, bcur, acc);
        bcur = bnx;
      }
    }
    {
      const bf16_t* pb = Whh + (size_t)n * H + koffB;
      v16bf bcur = *(const v16bf*)pb;
      #pragma unroll
      for (int kb = 0; kb < H; kb += 32) {
        v16bf bnx = bcur;
        if (kb + 32 < H) bnx = *(const v16bf*)(pb + kb + 32);
        v16bf a = mk16(&ldsH[rowA * H + kb + koffA]);
        acc = wmma_bf16(a, bcur, acc);
        bcur = bnx;
      }
    }
    if (lane < 16) {
      float bias = b_ih[n] + b_hh[n];
      for (int r = 0; r < 8; r++) gates[r * 2048 + n] = acc[r] + bias;
    }
  }
  __threadfence();
  __syncthreads();

  // P2: LSTM elementwise (torch gate order i,f,g,o)
  for (int i = tid; i < 8 * H; i += 256) {
    int b = i >> 9, j = i & (H - 1);
    float ig = gates[b * 2048 + j];
    float fg = gates[b * 2048 + 512 + j];
    float gg = gates[b * 2048 + 1024 + j];
    float og = gates[b * 2048 + 1536 + j];
    float cp = cws[b * H + j];
    float cn = sigf(fg) * cp + sigf(ig) * tanhf(gg);
    float hn = sigf(og) * tanhf(cn);
    cws[b * H + j] = cn;
    hws[b * H + j] = hn;
    ldsH[b * H + j] = (bf16_t)hn;
  }
  __threadfence();
  __syncthreads();

  // P3: Wh = h@W_hid^T + b_hid ; hsW[t] = h@W_mem^T + b_mem  (incremental attention keys)
  for (int strip = wv; strip < 64; strip += 8) {
    bool isMem = strip >= 32;
    int n = (strip & 31) * 16 + (lane & 15);
    const bf16_t* W = isMem ? Wmem : Whid;
    v8f acc; for (int r = 0; r < 8; r++) acc[r] = 0.f;
    const bf16_t* pb = W + (size_t)n * H + koffB;
    v16bf bcur = *(const v16bf*)pb;
    #pragma unroll
    for (int kb = 0; kb < H; kb += 32) {
      v16bf bnx = bcur;
      if (kb + 32 < H) bnx = *(const v16bf*)(pb + kb + 32);
      v16bf a = mk16(&ldsH[rowA * H + kb + koffA]);
      acc = wmma_bf16(a, bcur, acc);
      bcur = bnx;
    }
    if (lane < 16) {
      float bias = isMem ? b_mem[n] : b_hid[n];
      float* dst = isMem ? (hsW + (size_t)t * B * H) : Whb;
      for (int r = 0; r < 8; r++) dst[r * H + n] = acc[r] + bias;
    }
  }
  __threadfence();
  __syncthreads();

  // P4: attention scores sc[b][a] = v . tanh(Wh[b] + hsW[s0+a][b]) + v_b  (masked)
  const int s0 = (t > A) ? (t - A) : 0;
  const float vb = v_b[0];
  for (int p = wv; p < B * A; p += 8) {
    int b = p & 7, a = p >> 3;
    int pos = s0 + a;
    float part = 0.f;
    for (int k = lane; k < H; k += 32)
      part += v_w[k] * tanhf(Whb[b * H + k] + hsW[((size_t)pos * B + b) * H + k]);
    part = wred_sum(part);
    if (lane == 0) {
      bool masked = (pos >= t) || (n_tensor[b * L + pos] == VN - 1);
      ldsSc[b * 64 + a] = masked ? -1e20f : (part + vb);
    }
  }
  __syncthreads();

  // P5: softmax + log-softmax over window (one wave per batch row)
  {
    int b = wv;
    float m = -INFINITY;
    for (int a = lane; a < A; a += 32) m = fmaxf(m, ldsSc[b * 64 + a]);
    m = wred_max(m);
    float ss = 0.f;
    for (int a = lane; a < A; a += 32) ss += expf(ldsSc[b * 64 + a] - m);
    ss = wred_sum(ss);
    float lz = m + logf(ss);
    for (int a = lane; a < A; a += 32) {
      float sc = ldsSc[b * 64 + a];
      ldsAttn[b * 64 + a] = expf(sc - m) / ss;
      lgat[b * 64 + a] = sc - lz;
    }
  }
  __syncthreads();

  // P6: context + z = [context | h | h_parent]  (bf16 for the big GEMM)
  for (int i = tid; i < 8 * H; i += 256) {
    int b = i >> 9, k = i & (H - 1);
    float cx = 0.f;
    for (int a = 0; a < A; a++)
      cx += ldsAttn[b * 64 + a] * hs[((size_t)(s0 + a) * B + b) * H + k];
    ldsCtx[b * H + k] = cx;
    int pb = ldsIdx[16 + b];
    float hp = hs[((size_t)pb * B + b) * H + k];
    float hv = hws[b * H + k];
    zbuf[b * K3H + k]         = (bf16_t)cx;
    zbuf[b * K3H + H + k]     = (bf16_t)hv;
    zbuf[b * K3H + 2 * H + k] = (bf16_t)hp;
  }
  __syncthreads();

  // P7: st = log_sigmoid([context|h] . W_s + b_s)
  {
    int b = wv;
    float d = 0.f;
    for (int k = lane; k < 2 * H; k += 32)
      d += W_s[k] * ((k < H) ? ldsCtx[b * H + k] : hws[b * H + (k - H)]);
    d = wred_sum(d);
    if (lane == 0) {
      float x = d + b_s[0];
      stws[b] = fminf(x, 0.f) - log1pf(expf(-fabsf(x)));
    }
  }

  // P8: commit hs[t] = h (after h_parent gather)
  for (int i = tid; i < 8 * H; i += 256)
    hs[(size_t)t * B * H + i] = hws[i];
}

__global__ void __launch_bounds__(256)
k_gemm(char* __restrict__ ws, const float* __restrict__ b_g) {
  const int tid = threadIdx.x;
  const int wv = tid >> 5, lane = tid & 31;
  const bf16_t* WgT = (const bf16_t*)(ws + OFF_WGT);
  const bf16_t* z   = (const bf16_t*)(ws + OFF_ZBUF);
  float* logits     = (float*)(ws + OFF_LOGITS);

  const int n = (blockIdx.x * 8 + wv) * 16 + (lane & 15);
  const int koffA = (lane < 16) ? 0 : 16;
  const int koffB = (lane < 16) ? 0 : 16;
  const int rowA  = lane & 15;

  v8f acc; for (int r = 0; r < 8; r++) acc[r] = 0.f;
  const bf16_t* pa = z + (size_t)rowA * K3H + koffA;
  const bf16_t* pb = WgT + (size_t)n * K3H + koffB;

  // 1-deep software pipeline: issue next iteration's A/B loads before the
  // WMMA that consumes the current fragments, so L2 latency overlaps math.
  v16bf acur = mk16(pa);
  v16bf bcur = *(const v16bf*)pb;
  #pragma unroll
  for (int kb = 0; kb < K3H; kb += 32) {
    v16bf anx = acur, bnx = bcur;
    if (kb + 32 < K3H) {
      anx = mk16(pa + kb + 32);
      bnx = *(const v16bf*)(pb + kb + 32);
    }
    acc = wmma_bf16(acur, bcur, acc);
    acur = anx;
    bcur = bnx;
  }

  if (lane < 16) {  // lanes 0..15 hold batch rows 0..7 (padding rows live in lanes 16..31)
    bool valid = n < NG;
    float bias = valid ? b_g[n] : 0.f;
    for (int r = 0; r < 8; r++)
      logits[(size_t)r * NPAD + n] = valid ? (acc[r] + bias) : -1e30f;
  }
}

__global__ void __launch_bounds__(256)
k_reduce(char* __restrict__ ws, int t, const int* __restrict__ t_tensor,
         float* __restrict__ out) {
  __shared__ float sm[8];
  __shared__ int   si[8];
  __shared__ float ssh[8];
  __shared__ float bc;
  __shared__ int   bci;
  const int b = blockIdx.x;
  const int tid = threadIdx.x, wv = tid >> 5, lane = tid & 31;
  const float* logits = (const float*)(ws + OFF_LOGITS) + (size_t)b * NPAD;
  const float* stws   = (const float*)(ws + OFF_ST);
  const float* lgat   = (const float*)(ws + OFF_LOGATT);
  float* lossb        = (float*)(ws + OFF_LOSS);

  // pass 1: max + argmax (first occurrence on ties)
  float m = -INFINITY; int mi = 0;
  for (int n = tid; n < NG; n += 256) {
    float v = logits[n];
    if (v > m) { m = v; mi = n; }
  }
  for (int o = 16; o; o >>= 1) {
    float om = __shfl_xor(m, o, 32);
    int   oi = __shfl_xor(mi, o, 32);
    if (om > m || (om == m && oi < mi)) { m = om; mi = oi; }
  }
  if (lane == 0) { sm[wv] = m; si[wv] = mi; }
  __syncthreads();
  if (wv == 0) {
    float m2 = (lane < 8) ? sm[lane] : -INFINITY;
    int mi2  = (lane < 8) ? si[lane] : 0;
    for (int o = 4; o; o >>= 1) {
      float om = __shfl_xor(m2, o, 32);
      int   oi = __shfl_xor(mi2, o, 32);
      if (om > m2 || (om == m2 && oi < mi2)) { m2 = om; mi2 = oi; }
    }
    if (lane == 0) { bc = m2; bci = mi2; }
  }
  __syncthreads();
  const float gmax = bc; const int gidx = bci;

  // pass 2: sum exp
  float s = 0.f;
  for (int n = tid; n < NG; n += 256) s += expf(logits[n] - gmax);
  s = wred_sum(s);
  if (lane == 0) ssh[wv] = s;
  __syncthreads();

  if (tid == 0) {
    float tot = 0.f;
    for (int w = 0; w < 8; w++) tot += ssh[w];
    float logZ = gmax + logf(tot);
    float st = stws[b];
    float best = st + gmax - logZ;
    int pred = gidx;
    float q = -expm1f(st);
    q = fmaxf(q, 1e-18f);
    float lm = logf(q);
    for (int a = 0; a < A; a++) {
      float v = lm + lgat[b * 64 + a];
      if (v > best) { best = v; pred = NG + a; }
    }
    int tgt = t_tensor[b * L + t];
    float loss = -(st + logits[tgt] - logZ);
    lossb[t * B + b] = loss;
    out[1 + b * L + t] = (float)pred;
  }
}

__global__ void k_final(char* __restrict__ ws, float* __restrict__ out) {
  if (threadIdx.x == 0 && blockIdx.x == 0) {
    const float* lossb = (const float*)(ws + OFF_LOSS);
    float s = 0.f;
    for (int i = 0; i < L * B; i++) s += lossb[i];  // fixed order -> deterministic
    out[0] = s;
  }
}

// ------------------------- host entry -------------------------

extern "C" void kernel_launch(void* const* d_in, const int* in_sizes, int n_in,
                              void* d_out, int out_size, void* d_ws, size_t ws_size,
                              hipStream_t stream) {
  const int*   n_tensor = (const int*)d_in[0];
  const int*   t_tensor = (const int*)d_in[1];
  const int*   p_tensor = (const int*)d_in[2];
  const float* embN  = (const float*)d_in[3];
  const float* embT  = (const float*)d_in[4];
  const float* W_hid = (const float*)d_in[5];
  const float* b_hid = (const float*)d_in[6];
  const float* W_mem = (const float*)d_in[7];
  const float* b_mem = (const float*)d_in[8];
  const float* v_w   = (const float*)d_in[9];
  const float* v_b   = (const float*)d_in[10];
  const float* W_ih  = (const float*)d_in[11];
  const float* W_hh  = (const float*)d_in[12];
  const float* b_ih  = (const float*)d_in[13];
  const float* b_hh  = (const float*)d_in[14];
  const float* W_g   = (const float*)d_in[15];
  const float* b_g   = (const float*)d_in[16];
  const float* W_s   = (const float*)d_in[17];
  const float* b_s   = (const float*)d_in[18];
  char* ws   = (char*)d_ws;
  float* out = (float*)d_out;

  (void)in_sizes; (void)n_in; (void)out_size; (void)ws_size;

  // one-time (per launch / per graph replay) state zero + bf16 weight prep
  k_zero<<<dim3(512), dim3(256), 0, stream>>>(ws);
  k_cvt_small<<<dim3(2048), dim3(256), 0, stream>>>(ws, W_ih, W_hh, W_hid, W_mem);
  k_cvt_wg<<<dim3(NPAD), dim3(256), 0, stream>>>(ws, W_g);

  for (int t = 0; t < L; t++) {
    k_step<<<dim3(1), dim3(256), 0, stream>>>(ws, t, embN, embT,
                                              n_tensor, t_tensor, p_tensor,
                                              b_ih, b_hh, b_hid, b_mem,
                                              v_w, v_b, W_s, b_s);
    k_gemm<<<dim3(GEMM_BLOCKS), dim3(256), 0, stream>>>(ws, b_g);
    k_reduce<<<dim3(B), dim3(256), 0, stream>>>(ws, t, t_tensor, out);
  }
  k_final<<<dim3(1), dim3(1), 0, stream>>>(ws, out);
}
